// Quant_Linear_38182259261664
// MI455X (gfx1250) — 8-bit matrix operand order repaired and checked by reading, compiled, not run on hardware
//
#include <hip/hip_runtime.h>
#include <hip/hip_bf16.h>

#define M_DIM 8192
#define K_DIM 4096
#define N_DIM 4096

#define BM 128
#define BN 128
#define BK 128
#define LDR 144
#define NKITER (K_DIM / BK)

typedef __attribute__((ext_vector_type(8))) int v8i;
typedef int vi4 __attribute__((vector_size(16)));

#if defined(__has_builtin)
#if __has_builtin(__builtin_amdgcn_global_load_async_to_lds_b128)
#define HAVE_ASYNC_LDS 1
#endif
#if __has_builtin(__builtin_amdgcn_s_wait_asynccnt)
#define HAVE_WAIT_ASYNC 1
#endif
#endif

__device__ __forceinline__ void async_copy16(const signed char* g,
                                             signed char* l) {
#ifdef HAVE_ASYNC_LDS
  __builtin_amdgcn_global_load_async_to_lds_b128(
      (__attribute__((address_space(1))) vi4*)(g),
      (__attribute__((address_space(3))) vi4*)(l), 0, 0);
#else
  *(int4*)l = *(const int4*)g;
#endif
}

__device__ __forceinline__ void wait_async_copies() {
#ifdef HAVE_ASYNC_LDS
#ifdef HAVE_WAIT_ASYNC
  __builtin_amdgcn_s_wait_asynccnt(0);
#else
  asm volatile("s_wait_asynccnt 0" ::: "memory");
#endif
#endif
}

__global__ __launch_bounds__(256) void quant_w_kernel(
    const float* __restrict__ weight, const float* __restrict__ bias,
    const float* __restrict__ scale_x, signed char* __restrict__ wq,
    float* __restrict__ scale_out, int* __restrict__ bq) {
  const int row = blockIdx.x;
  const int tid = threadIdx.x;
  const float* wr = weight + (size_t)row * K_DIM;

  float amax = 0.f;
  for (int i = tid * 4; i < K_DIM; i += 256 * 4) {
    float4 v = *(const float4*)(wr + i);
    amax = fmaxf(amax, fmaxf(fmaxf(fabsf(v.x), fabsf(v.y)),
                             fmaxf(fabsf(v.z), fabsf(v.w))));
  }
  #pragma unroll
  for (int off = 16; off > 0; off >>= 1)
    amax = fmaxf(amax, __shfl_xor(amax, off, 32));

  __shared__ float red[8];
  if ((tid & 31) == 0) red[tid >> 5] = amax;
  __syncthreads();
  float m = red[0];
  #pragma unroll
  for (int j = 1; j < 8; ++j) m = fmaxf(m, red[j]);

  const float sw  = m * (1.0f / 127.0f);
  const float inv = (sw > 0.f) ? (1.0f / sw) : 0.f;

  for (int i = tid * 4; i < K_DIM; i += 256 * 4) {
    float4 v = *(const float4*)(wr + i);
    int q0 = min(127, max(-128, __float2int_rn(v.x * inv)));
    int q1 = min(127, max(-128, __float2int_rn(v.y * inv)));
    int q2 = min(127, max(-128, __float2int_rn(v.z * inv)));
    int q3 = min(127, max(-128, __float2int_rn(v.w * inv)));
    int packed = (q0 & 0xff) | ((q1 & 0xff) << 8) | ((q2 & 0xff) << 16)
               | ((q3 & 0xff) << 24);
    *(int*)(wq + (size_t)row * K_DIM + i) = packed;
  }

  if (tid == 0) {
    const float so = sw * scale_x[0];
    scale_out[row] = so;
    float bb = (so != 0.f) ? rintf(bias[row] / so) : 0.f;
    bb = fminf(fmaxf(bb, -2147483648.0f), 2147483520.0f);
    bq[row] = (int)(long long)bb;
  }
}

__global__ __launch_bounds__(256) void quant_x_kernel(
    const float* __restrict__ x, signed char* __restrict__ xq) {
  const size_t i = ((size_t)blockIdx.x * 256 + threadIdx.x) * 4;
  float4 v = *(const float4*)(x + i);
  int q0 = (int)v.x, q1 = (int)v.y, q2 = (int)v.z, q3 = (int)v.w;
  int packed = (q0 & 0xff) | ((q1 & 0xff) << 8) | ((q2 & 0xff) << 16)
             | ((q3 & 0xff) << 24);
  *(int*)(xq + i) = packed;
}

__global__ __launch_bounds__(256) void gemm_iu8_kernel(
    const signed char* __restrict__ xq, const signed char* __restrict__ wq,
    const float* __restrict__ scale_out, const int* __restrict__ bq,
    float* __restrict__ out) {
  __shared__ alignas(16) signed char As[2][BM * LDR];
  __shared__ alignas(16) signed char Bs[2][BN * LDR];

  const int tid  = threadIdx.x;
  const int bm   = blockIdx.y;
  const int bn   = blockIdx.x;
  const int lane = tid & 31;
  const int wave = tid >> 5;
  const int waveM = wave >> 1;
  const int waveN = wave & 1;
  const int m0 = waveM * 32;
  const int n0 = waveN * 64;
  const int half = lane >> 4;
  const int l    = lane & 15;

  const signed char* Ag = xq + (size_t)(bm * BM) * K_DIM;
  const signed char* Bg = wq + (size_t)(bn * BN) * K_DIM;

  const int r0 = tid >> 3;
  const int c0 = (tid & 7) * 16;

  v8i acc[2][4];
  const v8i vzero = {0, 0, 0, 0, 0, 0, 0, 0};
  #pragma unroll
  for (int tm = 0; tm < 2; ++tm)
    #pragma unroll
    for (int tn = 0; tn < 4; ++tn) acc[tm][tn] = vzero;

  #pragma unroll
  for (int i = 0; i < 4; ++i) {
    const int row = r0 + 32 * i;
    async_copy16(Ag + (size_t)row * K_DIM + c0, &As[0][row * LDR + c0]);
    async_copy16(Bg + (size_t)row * K_DIM + c0, &Bs[0][row * LDR + c0]);
  }
  wait_async_copies();
  __syncthreads();

  for (int kt = 0; kt < NKITER; ++kt) {
    const int s = kt & 1;

    if (kt + 1 < NKITER) {
      const int k1 = (kt + 1) * BK;
      #pragma unroll
      for (int i = 0; i < 4; ++i) {
        const int row = r0 + 32 * i;
        async_copy16(Ag + (size_t)row * K_DIM + k1 + c0,
                     &As[s ^ 1][row * LDR + c0]);
        async_copy16(Bg + (size_t)row * K_DIM + k1 + c0,
                     &Bs[s ^ 1][row * LDR + c0]);
      }
    }
    if (kt + 2 < NKITER) {
      const int k2 = (kt + 2) * BK;
      __builtin_prefetch(Ag + (size_t)r0 * K_DIM + k2 + c0, 0, 1);
      __builtin_prefetch(Bg + (size_t)r0 * K_DIM + k2 + c0, 0, 1);
    }

    #pragma unroll
    for (int kk = 0; kk < 2; ++kk) {
      #pragma unroll
      for (int tm = 0; tm < 2; ++tm) {
        v8i a;
        const signed char* abase =
            &As[s][(m0 + tm * 16 + l) * LDR + kk * 64 + half * 8];
        #pragma unroll
        for (int p = 0; p < 4; ++p) {
          int2 t = *(const int2*)(abase + p * 16);
          a[2 * p] = t.x;
          a[2 * p + 1] = t.y;
        }
        #pragma unroll
        for (int tn = 0; tn < 4; ++tn) {
          const signed char* bbase =
              &Bs[s][(n0 + tn * 16 + l) * LDR + kk * 64 + half * 8];
          v8i b;
          #pragma unroll
          for (int p = 0; p < 4; ++p) {
            int2 t = *(const int2*)(bbase + p * 16);
            b[2 * p] = t.x;
            b[2 * p + 1] = t.y;
          }
          acc[tm][tn] = __builtin_amdgcn_wmma_i32_16x16x64_iu8(
              true, a, true, b, acc[tm][tn], false, false);
        }
      }
    }

    wait_async_copies();
    __syncthreads();
  }

  #pragma unroll
  for (int tn = 0; tn < 4; ++tn) {
    const int col_g = bn * BN + n0 + tn * 16 + l;
    const int   bqv = bq[col_g];
    const float sc  = scale_out[col_g];
    #pragma unroll
    for (int tm = 0; tm < 2; ++tm) {
      #pragma unroll
      for (int r = 0; r < 8; ++r) {
        const int row_g = bm * BM + m0 + tm * 16 + r + half * 8;
        out[(size_t)row_g * N_DIM + col_g] =
            (float)(acc[tm][tn][r] + bqv) * sc;
      }
    }
  }
}

extern "C" void kernel_launch(void* const* d_in, const int* in_sizes, int n_in,
                              void* d_out, int out_size, void* d_ws,
                              size_t ws_size, hipStream_t stream) {
  const float* x_q     = (const float*)d_in[0];
  const float* scale_x = (const float*)d_in[1];
  const float* weight  = (const float*)d_in[2];
  const float* bias    = (const float*)d_in[3];
  float* out           = (float*)d_out;

  signed char* wq = (signed char*)d_ws;
  signed char* xq = wq + (size_t)N_DIM * K_DIM;
  float* scale_out = (float*)(xq + (size_t)M_DIM * K_DIM);
  int* bq = (int*)(scale_out + N_DIM);

  (void)in_sizes; (void)n_in; (void)out_size; (void)ws_size;

  quant_w_kernel<<<N_DIM, 256, 0, stream>>>(weight, bias, scale_x, wq,
                                            scale_out, bq);

  const size_t nx = (size_t)M_DIM * K_DIM / 4;
  quant_x_kernel<<<(unsigned)(nx / 256), 256, 0, stream>>>(x_q, xq);

  dim3 grid(N_DIM / BN, M_DIM / BM);
  gemm_iu8_kernel<<<grid, 256, 0, stream>>>(xq, wq, scale_out, bq, out);
}
